// PLELayer_57251914056207
// MI455X (gfx1250) — matrix operand order repaired and checked by reading, not compiled, not run on hardware
//
#include <hip/hip_runtime.h>
#include <hip/hip_bf16.h>
#include <stdint.h>

typedef __attribute__((ext_vector_type(16))) __bf16 v16bf;
typedef __attribute__((ext_vector_type(8)))  float  v8f;

#define D_DIM 256
#define U_DIM 256
#define BM 32
#define A_LD 264            // padded LDS row stride in bf16 elements (bank-conflict free)
#define NEXP 6              // 4 task experts + 2 shared experts per level
#define NFRAG 128           // 8 K-tiles * 16 N-tiles per 256x256 weight
#define FRAG_ELEMS 512      // 32x16 bf16 fragment

__device__ __forceinline__ unsigned short f2bf(float f) {
  union { float f; uint32_t u; } v; v.f = f;
  uint32_t u = v.u;
  return (unsigned short)((u + 0x7FFFu + ((u >> 16) & 1u)) >> 16);  // RNE
}

// ---------------------------------------------------------------------------
// Repack fp32 expert weights [D,U] into bf16 WMMA B-fragments.
// Fragment layout (dense 16x16x32 bf16 B, 32x16 K-by-N):
//   lane l (0..31), vgpr v (0..7), half h: K = 16*(l>>4) + 2v + h, N = l&15
// Each lane's 16 bf16 (32B) are contiguous -> two coalesced b128 loads/frag.
// ---------------------------------------------------------------------------
__global__ void pack_weights_kernel(const float* __restrict__ we_task,
                                    const float* __restrict__ we_share,
                                    unsigned short* __restrict__ packedW) {
  int i = blockIdx.x * blockDim.x + threadIdx.x;
  if (i >= 2 * NEXP * NFRAG * FRAG_ELEMS) return;
  int h    = i & 1;
  int v    = (i >> 1) & 7;
  int lane = (i >> 4) & 31;
  int frag = (i >> 9) & 127;
  int le   = i >> 16;            // 0..11 = lvl*6 + e
  int e    = le % NEXP;
  int lvl  = le / NEXP;
  int kt = frag >> 4, ntile = frag & 15;
  int K = kt * 32 + (lane >> 4) * 16 + 2 * v + h;
  int N = ntile * 16 + (lane & 15);
  float w = (e < 4)
      ? we_task [(size_t)(lvl * 4 + e)       * (D_DIM * U_DIM) + (size_t)K * U_DIM + N]
      : we_share[(size_t)(lvl * 2 + (e - 4)) * (D_DIM * U_DIM) + (size_t)K * U_DIM + N];
  packedW[i] = f2bf(w);
}

// Pack gate weights, N-padded to 16 with zeros, same fragment layout.
// packedGt: [L*T][8 kt][512]  (task gates, D x 4 used)
// packedGs: [8 kt][512]       (shared gate, D x 6 used, level 0 only)
__global__ void pack_gates_kernel(const float* __restrict__ wg_task,
                                  const float* __restrict__ wg_share,
                                  unsigned short* __restrict__ packedGt,
                                  unsigned short* __restrict__ packedGs) {
  int i = blockIdx.x * blockDim.x + threadIdx.x;
  if (i >= (2 * 2 * 8 + 8) * FRAG_ELEMS) return;
  int h    = i & 1;
  int v    = (i >> 1) & 7;
  int lane = (i >> 4) & 31;
  int frag = i >> 9;             // 0..31 task (lt=frag>>3, kt=frag&7), 32..39 share
  int N = lane & 15;
  if (frag < 32) {
    int lt = frag >> 3, kt = frag & 7;
    int K = kt * 32 + (lane >> 4) * 16 + 2 * v + h;
    float w = (N < 4) ? wg_task[(size_t)lt * (D_DIM * 4) + (size_t)K * 4 + N] : 0.0f;
    packedGt[(size_t)frag * FRAG_ELEMS + (i & 511)] = f2bf(w);
  } else {
    int kt = frag - 32;
    int K = kt * 32 + (lane >> 4) * 16 + 2 * v + h;
    float w = (N < 6) ? wg_share[(size_t)K * 6 + N] : 0.0f;
    packedGs[(size_t)kt * FRAG_ELEMS + (i & 511)] = f2bf(w);
  }
}

// ---------------------------------------------------------------------------
// Fused PLE level. Block: 256 threads = 8 waves; tile 32 rows x 256 cols.
// Phase 1: stage A tiles to LDS (bf16). Phase 2: gate logits via WMMA
// (waves 0-3 task gates, waves 4-5 shared gates), softmax. Phase 3: 6 expert
// GEMMs (v_wmma_f32_16x16x32_bf16, wave tile 16x64) with bias+ReLU+gate-
// weighted combine held in registers. LAST=1 drops the shared branch and
// stores fp32 to d_out; LAST=0 stores bf16 intermediates.
// ---------------------------------------------------------------------------
template <int LAST, int BF16SRC>
__global__ __launch_bounds__(256)
void ple_level_kernel(int lvl,
                      const void* __restrict__ srcT0,
                      const void* __restrict__ srcT1,
                      const void* __restrict__ srcS,
                      const unsigned short* __restrict__ packedW,
                      const unsigned short* __restrict__ packedGt,
                      const unsigned short* __restrict__ packedGs,
                      const float* __restrict__ bg_task,
                      const float* __restrict__ bg_share,
                      const float* __restrict__ be_task,
                      const float* __restrict__ be_share,
                      unsigned short* __restrict__ outT0_bf,
                      unsigned short* __restrict__ outT1_bf,
                      unsigned short* __restrict__ outS_bf,
                      float* __restrict__ outT0_f,
                      float* __restrict__ outT1_f) {
  __shared__ unsigned short As[3][BM * A_LD];
  __shared__ float lgt[2][BM][4];
  __shared__ float gt [2][BM][4];
  __shared__ float lgs[BM][6];
  __shared__ float gs [BM][6];

  const int tid = threadIdx.x;
  const size_t rows0 = (size_t)blockIdx.x * BM;

  const int sel1 = (srcT1 == srcT0) ? 0 : 1;   // level 0: all inputs alias x
  const int selS = (srcS  == srcT0) ? 0 : 2;

  const int wid = tid >> 5, lane = tid & 31;
  const int lg = lane >> 4, ln = lane & 15;

  // ---- stage A tiles into LDS as bf16 ----
  const void* srcs[3] = { srcT0, srcT1, srcS };
  for (int s = 0; s < 3; ++s) {
    if (s == 1 && sel1 == 0) continue;
    if (s == 2 && selS == 0) continue;
    for (int i = tid; i < BM * (D_DIM / 4); i += 256) {
      int r  = i >> 6;
      int c4 = (i & 63) * 4;
      unsigned short* dst = &As[s][r * A_LD + c4];
      size_t off = (rows0 + r) * D_DIM + c4;
      if (BF16SRC) {
        *(uint2*)dst = *(const uint2*)((const unsigned short*)srcs[s] + off);
      } else {
        float4 f = *(const float4*)((const float*)srcs[s] + off);
        dst[0] = f2bf(f.x); dst[1] = f2bf(f.y);
        dst[2] = f2bf(f.z); dst[3] = f2bf(f.w);
      }
    }
  }
  __syncthreads();

  // ---- gate logits via WMMA ----
  // waves 0..3: task gates (t = wid>>1, m-tile = wid&1) -> 32x16 logits (cols 0..3 valid)
  // waves 4..5 (level 0): shared gates (m-tile = wid-4)  -> cols 0..5 valid
  if (wid < 4 || (!LAST && wid < 6)) {
    const int isShare = (wid >= 4);
    const int t  = isShare ? 0 : (wid >> 1);
    const int mt = isShare ? (wid - 4) : (wid & 1);
    const int sel = isShare ? selS : (t ? sel1 : 0);
    const unsigned short* Arow = &As[sel][(mt * 16 + ln) * A_LD];
    const unsigned short* gw = isShare
        ? packedGs
        : packedGt + (size_t)(lvl * 2 + t) * 8 * FRAG_ELEMS;
    v8f acc = {};
#pragma unroll
    for (int kt = 0; kt < 8; ++kt) {
      union { uint32_t u[8]; v16bf v; } af;
      const int kb = kt * 32 + lg * 16;
#pragma unroll
      for (int v = 0; v < 8; ++v)
        af.u[v] = *(const uint32_t*)(Arow + kb + v * 2);
      union { uint4 q[2]; v16bf v; } bfv;
      const uint4* p = (const uint4*)(gw + (size_t)kt * FRAG_ELEMS + lane * 16);
      bfv.q[0] = p[0]; bfv.q[1] = p[1];
      acc = __builtin_amdgcn_wmma_f32_16x16x32_bf16(
          false, af.v, false, bfv.v, (short)0, acc, false, false);
    }
    // D layout: N = ln, M = r8 + 8*lg
    if (!isShare && ln < 4) {
      const float bias = bg_task[(lvl * 2 + t) * 4 + ln];
#pragma unroll
      for (int r8 = 0; r8 < 8; ++r8)
        lgt[t][mt * 16 + lg * 8 + r8][ln] = acc[r8] + bias;
    }
    if (isShare && ln < 6) {
      const float bias = bg_share[ln];
#pragma unroll
      for (int r8 = 0; r8 < 8; ++r8)
        lgs[mt * 16 + lg * 8 + r8][ln] = acc[r8] + bias;
    }
  }
  __syncthreads();

  // ---- softmax ----
  {
    int t = tid >> 7, r = (tid >> 2) & 31, g = tid & 3;
    float l0 = lgt[t][r][0], l1 = lgt[t][r][1], l2 = lgt[t][r][2], l3 = lgt[t][r][3];
    float m  = fmaxf(fmaxf(l0, l1), fmaxf(l2, l3));
    float e0 = __expf(l0 - m), e1 = __expf(l1 - m), e2 = __expf(l2 - m), e3 = __expf(l3 - m);
    float inv = 1.0f / (e0 + e1 + e2 + e3);
    float eg = (g == 0) ? e0 : (g == 1) ? e1 : (g == 2) ? e2 : e3;
    gt[t][r][g] = eg * inv;
  }
  if (!LAST && tid < BM * 6) {
    int r = tid / 6, g = tid % 6;
    float l[6], m = -3.4e38f;
#pragma unroll
    for (int j = 0; j < 6; ++j) { l[j] = lgs[r][j]; m = fmaxf(m, l[j]); }
    float s = 0.f;
#pragma unroll
    for (int j = 0; j < 6; ++j) { l[j] = __expf(l[j] - m); s += l[j]; }
    gs[r][g] = l[g] / s;
  }
  __syncthreads();

  // ---- expert GEMMs + fused combine ----
  const int wm = wid >> 2, wn = wid & 3;     // 2 waves in M, 4 in N

  v8f comb0[4], comb1[4], comb2[4];
  { v8f z = {};
#pragma unroll
    for (int nt = 0; nt < 4; ++nt) {
      comb0[nt] = z; comb1[nt] = z;
      if (!LAST) comb2[nt] = z;
    } }

  for (int e = 0; e < NEXP; ++e) {
    const int sel = (e < 4) ? ((e >> 1) ? sel1 : 0) : selS;
    const unsigned short* Arow = &As[sel][(wm * 16 + ln) * A_LD];
    const unsigned short* wbase =
        packedW + (size_t)(lvl * NEXP + e) * NFRAG * FRAG_ELEMS;

    v8f acc[4];
    { v8f z = {}; acc[0] = z; acc[1] = z; acc[2] = z; acc[3] = z; }

#pragma unroll
    for (int kt = 0; kt < 8; ++kt) {
      // A fragment 16x32 bf16 from LDS (lane m = ln, group lg:
      // vgpr v holds K = kt*32 + 16*lg + 2*v + h)
      union { uint32_t u[8]; v16bf v; } af;
      const int kb = kt * 32 + lg * 16;
#pragma unroll
      for (int v = 0; v < 8; ++v)
        af.u[v] = *(const uint32_t*)(Arow + kb + v * 2);
#pragma unroll
      for (int nt = 0; nt < 4; ++nt) {
        const int frag = kt * 16 + wn * 4 + nt;
        union { uint4 q[2]; v16bf v; } bfv;
        const uint4* p = (const uint4*)(wbase + (size_t)frag * FRAG_ELEMS + lane * 16);
        bfv.q[0] = p[0]; bfv.q[1] = p[1];
        acc[nt] = __builtin_amdgcn_wmma_f32_16x16x32_bf16(
            false, af.v, false, bfv.v, (short)0, acc[nt], false, false);
      }
    }

    // gate weights for this expert's 8 rows (hoisted out of the nt loop)
    float gv0[8], gv1[8], gv2[8];
#pragma unroll
    for (int r8 = 0; r8 < 8; ++r8) {
      const int rowT = wm * 16 + lg * 8 + r8;
      if (e < 4) {
        const int t = e >> 1, gi = e & 1;
        gv0[r8] = (t == 0) ? gt[0][rowT][gi] : 0.0f;
        gv1[r8] = (t == 1) ? gt[1][rowT][gi] : 0.0f;
        if (!LAST) gv2[r8] = gs[rowT][e];
      } else {
        const int gi = 2 + (e - 4);
        gv0[r8] = gt[0][rowT][gi];
        gv1[r8] = gt[1][rowT][gi];
        if (!LAST) gv2[r8] = gs[rowT][4 + (e - 4)];
      }
    }

    // bias + ReLU + gate-weighted accumulation (C layout: N=ln, M=r8+8*lg)
    const float* bias = (e < 4) ? be_task  + (size_t)(lvl * 4 + e)       * U_DIM
                                : be_share + (size_t)(lvl * 2 + (e - 4)) * U_DIM;
#pragma unroll
    for (int nt = 0; nt < 4; ++nt) {
      const float b = bias[wn * 64 + nt * 16 + ln];
#pragma unroll
      for (int r8 = 0; r8 < 8; ++r8) {
        float val = fmaxf(acc[nt][r8] + b, 0.0f);
        comb0[nt][r8] += gv0[r8] * val;
        comb1[nt][r8] += gv1[r8] * val;
        if (!LAST) comb2[nt][r8] += gv2[r8] * val;
      }
    }
  }

  // ---- store combined outputs ----
#pragma unroll
  for (int nt = 0; nt < 4; ++nt) {
    const int col = wn * 64 + nt * 16 + ln;
#pragma unroll
    for (int r8 = 0; r8 < 8; ++r8) {
      const size_t row = rows0 + wm * 16 + lg * 8 + r8;
      const size_t idx = row * U_DIM + col;
      if (LAST) {
        outT0_f[idx] = comb0[nt][r8];
        outT1_f[idx] = comb1[nt][r8];
      } else {
        outT0_bf[idx] = f2bf(comb0[nt][r8]);
        outT1_bf[idx] = f2bf(comb1[nt][r8]);
        outS_bf[idx]  = f2bf(comb2[nt][r8]);
      }
    }
  }
}

extern "C" void kernel_launch(void* const* d_in, const int* in_sizes, int n_in,
                              void* d_out, int out_size, void* d_ws, size_t ws_size,
                              hipStream_t stream) {
  const float* x        = (const float*)d_in[0];
  const float* we_task  = (const float*)d_in[1];
  const float* be_task  = (const float*)d_in[2];
  const float* we_share = (const float*)d_in[3];
  const float* be_share = (const float*)d_in[4];
  const float* wg_task  = (const float*)d_in[5];
  const float* bg_task  = (const float*)d_in[6];
  const float* wg_share = (const float*)d_in[7];
  const float* bg_share = (const float*)d_in[8];

  const int B = in_sizes[0] / D_DIM;       // 65536

  // workspace layout: [packedW | packedGt | packedGs | O0 | O1 | OS]
  unsigned short* packedW  = (unsigned short*)d_ws;
  unsigned short* packedGt = (unsigned short*)((char*)d_ws + 0x180000);
  unsigned short* packedGs = (unsigned short*)((char*)d_ws + 0x190000);
  const size_t oElems = (size_t)B * U_DIM;
  unsigned short* O0 = (unsigned short*)((char*)d_ws + (2u << 20));
  unsigned short* O1 = O0 + oElems;
  unsigned short* OS = O1 + oElems;

  const int packN = 2 * NEXP * NFRAG * FRAG_ELEMS;
  pack_weights_kernel<<<(packN + 255) / 256, 256, 0, stream>>>(we_task, we_share, packedW);
  const int packG = (2 * 2 * 8 + 8) * FRAG_ELEMS;
  pack_gates_kernel<<<(packG + 255) / 256, 256, 0, stream>>>(wg_task, wg_share,
                                                             packedGt, packedGs);

  // level 0: all three branch inputs alias x (fp32), writes bf16 intermediates
  ple_level_kernel<0, 0><<<B / BM, 256, 0, stream>>>(
      0, x, x, x, packedW, packedGt, packedGs,
      bg_task, bg_share, be_task, be_share,
      O0, O1, OS, nullptr, nullptr);

  // level 1: branch inputs are level-0 outputs (bf16), writes fp32 [T,B,U]
  float* out = (float*)d_out;
  ple_level_kernel<1, 1><<<B / BM, 256, 0, stream>>>(
      1, O0, O1, OS, packedW, packedGt, packedGs,
      bg_task, bg_share, be_task, be_share,
      nullptr, nullptr, nullptr, out, out + oElems);
}
